// FasterRCNN_1374389534724
// MI455X (gfx1250) — matrix operand order repaired and checked by reading, not compiled, not run on hardware
//
#include <hip/hip_runtime.h>
#include <hip/hip_bf16.h>
#include <math.h>

// ---------------------------------------------------------------------------
// Faster R-CNN head for MI455X (gfx1250, wave32, WMMA + TDM).
// FC1 (1024x25088x4096) / FC2 (1024x4096x4096) run on v_wmma_f32_16x16x32_bf16,
// A tiles DMA'd into LDS by the Tensor Data Mover (tensor_load_to_lds with
// LDS padding fields), B weights software-pipelined through registers with
// packed fp32->bf16 conversion. f32 accumulation throughout.
// ---------------------------------------------------------------------------

typedef __bf16 bf16;
typedef __attribute__((ext_vector_type(16))) __bf16 v16bf;
typedef __attribute__((ext_vector_type(8)))  __bf16 v8bf;
typedef __attribute__((ext_vector_type(2)))  __bf16 v2bf;
typedef __attribute__((ext_vector_type(8)))  float  v8f;
typedef __attribute__((ext_vector_type(4)))  unsigned int v4u;
typedef __attribute__((ext_vector_type(8)))  int v8i_t;
typedef __attribute__((ext_vector_type(4)))  int v4i_t;

#define NEGV (-1000000000.0f)

#define BATCH 2
#define CH    512
#define FH    38
#define FW    50
#define NPROP 2000
#define NSORT 2048
#define RKEEP 512
#define NCLS  21
#define DPOOL 25088   // 512 * 49
#define HID   4096
#define MROWS 1024    // BATCH * RKEEP
#define MAXDET 300

// ---------------------------------------------------------------------------
// 1) RPN NMS per image
// ---------------------------------------------------------------------------
__global__ __launch_bounds__(1024) void rpn_nms_kernel(
    const float* __restrict__ proposals, const float* __restrict__ obj,
    float* __restrict__ rois, int* __restrict__ roi_valid)
{
  __shared__ float ss[NSORT];
  __shared__ int   si[NSORT];
  __shared__ float bx[NSORT][4];
  __shared__ unsigned char keep[NSORT];
  __shared__ int   sel[RKEEP];
  __shared__ unsigned char selv[RKEEP];

  const int img = blockIdx.x;
  const int tid = threadIdx.x;
  const float* P = proposals + (size_t)img * NPROP * 4;
  const float* S = obj + (size_t)img * NPROP;

  for (int i = tid; i < NSORT; i += 1024) {
    ss[i] = (i < NPROP) ? S[i] : NEGV;
    si[i] = i;
  }
  __syncthreads();

  for (int k = 2; k <= NSORT; k <<= 1) {
    for (int j = k >> 1; j > 0; j >>= 1) {
      for (int i = tid; i < NSORT; i += 1024) {
        int p = i ^ j;
        if (p > i) {
          bool dir = ((i & k) == 0);
          float a = ss[i], b = ss[p];
          bool sw = dir ? (a < b) : (a > b);
          if (sw) { ss[i] = b; ss[p] = a; int t = si[i]; si[i] = si[p]; si[p] = t; }
        }
      }
      __syncthreads();
    }
  }

  for (int i = tid; i < NSORT; i += 1024) {
    int o = si[i];
    if (o < NPROP) {
      float x = P[o*4+0], y = P[o*4+1], w = P[o*4+2], h = P[o*4+3];
      bx[i][0] = x; bx[i][1] = y; bx[i][2] = x + w; bx[i][3] = y + h;
      keep[i] = 1;
    } else {
      bx[i][0] = bx[i][1] = bx[i][2] = bx[i][3] = 0.f;
      keep[i] = 0;
    }
  }
  __syncthreads();

  for (int i = 0; i < NPROP; ++i) {
    if (keep[i]) {
      float x1 = bx[i][0], y1 = bx[i][1], x2 = bx[i][2], y2 = bx[i][3];
      float ai = (x2 - x1) * (y2 - y1);
      for (int jj = i + 1 + tid; jj < NSORT; jj += 1024) {
        if (keep[jj]) {
          float u1 = fmaxf(x1, bx[jj][0]), v1 = fmaxf(y1, bx[jj][1]);
          float u2 = fminf(x2, bx[jj][2]), v2 = fminf(y2, bx[jj][3]);
          float iw = fmaxf(u2 - u1, 0.f), ih = fmaxf(v2 - v1, 0.f);
          float inter = iw * ih;
          float aj = (bx[jj][2] - bx[jj][0]) * (bx[jj][3] - bx[jj][1]);
          float iou = inter / fmaxf(ai + aj - inter, 1e-6f);
          if (iou > 0.7f) keep[jj] = 0;
        }
      }
    }
    __syncthreads();
  }

  if (tid == 0) {
    int cnt = 0;
    for (int i = 0; i < NSORT && cnt < RKEEP; ++i)
      if (keep[i]) { sel[cnt] = i; selv[cnt] = 1; ++cnt; }
    for (int i = 0; i < NSORT && cnt < RKEEP; ++i)
      if (!keep[i] && si[i] < NPROP) { sel[cnt] = i; selv[cnt] = 0; ++cnt; }
  }
  __syncthreads();

  for (int i = tid; i < RKEEP; i += 1024) {
    int o = si[sel[i]];
    rois[((size_t)img*RKEEP + i)*4 + 0] = P[o*4+0];
    rois[((size_t)img*RKEEP + i)*4 + 1] = P[o*4+1];
    rois[((size_t)img*RKEEP + i)*4 + 2] = P[o*4+2];
    rois[((size_t)img*RKEEP + i)*4 + 3] = P[o*4+3];
    roi_valid[(size_t)img*RKEEP + i] = selv[i];
  }
}

// ---------------------------------------------------------------------------
// 2) ROI max-pool -> bf16 activations
// ---------------------------------------------------------------------------
__global__ void roi_pool_kernel(const float* __restrict__ feat,
                                const float* __restrict__ rois,
                                bf16* __restrict__ pooled)
{
  int idx = blockIdx.x * blockDim.x + threadIdx.x;
  if (idx >= BATCH * RKEEP * CH) return;
  int c   = idx % CH;
  int r   = (idx / CH) % RKEEP;
  int img = idx / (CH * RKEEP);

  const float* R = rois + ((size_t)img * RKEEP + r) * 4;
  float x = R[0], y = R[1], w = R[2], h = R[3];
  float x1 = rintf(x / 16.f),       y1 = rintf(y / 16.f);
  float x2 = rintf((x + w) / 16.f), y2 = rintf((y + h) / 16.f);
  float rw = fmaxf(x2 - x1 + 1.f, 1.f), rh = fmaxf(y2 - y1 + 1.f, 1.f);

  int px[14], py[14];
  #pragma unroll
  for (int t = 0; t < 14; ++t) {
    float g = (float)(t >> 1) + ((t & 1) ? 0.75f : 0.25f);
    float fx = x1 + g * (rw / 7.f);
    float fy = y1 + g * (rh / 7.f);
    int ix = (int)floorf(fx); ix = ix < 0 ? 0 : (ix > FW - 1 ? FW - 1 : ix);
    int iy = (int)floorf(fy); iy = iy < 0 ? 0 : (iy > FH - 1 ? FH - 1 : iy);
    px[t] = ix; py[t] = iy;
  }

  const float* F = feat + ((size_t)img * CH + c) * (FH * FW);
  bf16* out = pooled + (size_t)(img * RKEEP + r) * DPOOL + c * 49;
  for (int oy = 0; oy < 7; ++oy)
    for (int ox = 0; ox < 7; ++ox) {
      float m =    F[py[2*oy  ] * FW + px[2*ox  ]];
      m = fmaxf(m, F[py[2*oy  ] * FW + px[2*ox+1]]);
      m = fmaxf(m, F[py[2*oy+1] * FW + px[2*ox  ]]);
      m = fmaxf(m, F[py[2*oy+1] * FW + px[2*ox+1]]);
      out[oy * 7 + ox] = (bf16)m;
    }
}

// ---------------------------------------------------------------------------
// 3) bf16 WMMA GEMM: C[M,N] = relu(A(bf16) * B(fp32->bf16) + bias)
//    block 256 thr (8 waves), tile 128x256, waves 2x4 each own 64x64
//    (4x4 wmma accumulators). A tile via TDM (tensor_load_to_lds) when
//    available, B tile software-pipelined through registers.
// ---------------------------------------------------------------------------
__global__ __launch_bounds__(256) void gemm_bf16_kernel(
    const bf16* __restrict__ A, const float* __restrict__ Bw,
    const float* __restrict__ bias, bf16* __restrict__ C,
    int K, int N, int relu)
{
  __shared__ __align__(16) bf16 lA[128 * 40];  // row-major [m][k], pad 40
  __shared__ __align__(16) bf16 lB[256 * 40];  // col-major [n][k], pad 40

  const int tid   = threadIdx.x;
  const int lane  = tid & 31;
  const int wave  = tid >> 5;
  const int wm    = wave >> 2;           // 0..1 -> M offset wm*64
  const int wn    = wave & 3;            // 0..3 -> N offset wn*64
  const int row16 = lane & 15;
  const int khalf = lane >> 4;
  const int mTile = blockIdx.y * 128;
  const int nTile = blockIdx.x * 256;

  v8f acc[4][4];
  #pragma unroll
  for (int i = 0; i < 4; ++i)
    #pragma unroll
    for (int j = 0; j < 4; ++j)
      #pragma unroll
      for (int r = 0; r < 8; ++r) acc[i][j][r] = 0.f;

  // B staging: thread covers k rows {k0,k0+1}, n cols n0..n0+15
  const int k0 = (tid >> 4) * 2;
  const int n0 = (tid & 15) * 16;
  float4 breg[8];

  // preload first B tile
  #pragma unroll
  for (int kk = 0; kk < 2; ++kk)
    #pragma unroll
    for (int u = 0; u < 4; ++u)
      breg[kk * 4 + u] = *(const float4*)(Bw + (size_t)(k0 + kk) * N + nTile + n0 + u * 4);

  union V16 { v16bf v; v8bf h[2]; };

  for (int kb = 0; kb < K; kb += 32) {
    __syncthreads();  // previous iteration's LDS consumers are done

    // ---- stage A tile (128 x 32 bf16) ----
#if defined(__gfx1250__) && __has_builtin(__builtin_amdgcn_tensor_load_to_lds)
    if (tid < 32) {  // one TDM op, issued by wave 0
      unsigned ldsA = (unsigned)(size_t)(void*)lA;
      unsigned long long ga =
          (unsigned long long)(size_t)(A + (size_t)mTile * K + kb);
      v4u g0;
      g0[0] = 1u;                                   // count=1 (valid D#)
      g0[1] = ldsA;                                 // lds_addr
      g0[2] = (unsigned)ga;                         // global_addr[31:0]
      g0[3] = (unsigned)((ga >> 32) & 0x1FFFFFFull) // global_addr[56:32]
              | (2u << 30);                         // type=2 (image)
      v8i_t g1;
      g1[0] = (1 << 16)      // data_size=1 -> 2 bytes
            | (1 << 20)      // pad_enable
            | (3 << 22)      // pad_interval: 16 DWORDs (64B) between pads
            | (3 << 25);     // pad_amount: 4 DWORDs (16B) -> LDS stride 80B
      g1[1] = (int)((unsigned)(K & 0xFFFF) << 16);          // tensor_dim0 lo
      g1[2] = (int)(((unsigned)K >> 16) & 0xFFFFu)          // tensor_dim0 hi
            | (int)((MROWS & 0xFFFF) << 16);                // tensor_dim1 lo
      g1[3] = 0                                             // tensor_dim1 hi
            | (32 << 16);                                   // tile_dim0 = 32
      g1[4] = 128;                                          // tile_dim1 = 128
      g1[5] = K;                                            // dim0_stride lo
      g1[6] = 0;
      g1[7] = 0;
      v4i_t z4 = {0, 0, 0, 0};
      v8i_t z8 = {0, 0, 0, 0, 0, 0, 0, 0};
      __builtin_amdgcn_tensor_load_to_lds(g0, g1, z4, z4, z8, 0);
    }
#else
    {
      const int arow = tid >> 1, ahalf = tid & 1;
      const v8bf* src =
          (const v8bf*)(A + (size_t)(mTile + arow) * K + kb + ahalf * 16);
      v8bf lo = src[0], hi = src[1];
      *(v8bf*)(lA + arow * 40 + ahalf * 16)     = lo;
      *(v8bf*)(lA + arow * 40 + ahalf * 16 + 8) = hi;
    }
#endif

    // ---- stage B tile from registers (fp32 -> bf16, transpose to [n][k]) ----
    #pragma unroll
    for (int u = 0; u < 4; ++u) {
      const float* f0 = (const float*)&breg[u];
      const float* f1 = (const float*)&breg[4 + u];
      #pragma unroll
      for (int e = 0; e < 4; ++e) {
        int n = n0 + u * 4 + e;
        v2bf pk; pk[0] = (bf16)f0[e]; pk[1] = (bf16)f1[e];
        *(v2bf*)(lB + n * 40 + k0) = pk;  // 4B store, k0 even
      }
    }

    // ---- issue global loads for next B tile (hidden behind WMMAs) ----
    if (kb + 32 < K) {
      #pragma unroll
      for (int kk = 0; kk < 2; ++kk)
        #pragma unroll
        for (int u = 0; u < 4; ++u)
          breg[kk * 4 + u] = *(const float4*)(
              Bw + (size_t)(kb + 32 + k0 + kk) * N + nTile + n0 + u * 4);
    }

#if defined(__gfx1250__) && __has_builtin(__builtin_amdgcn_tensor_load_to_lds)
    __builtin_amdgcn_s_wait_tensorcnt(0);
#endif
    __syncthreads();

    // ---- fragments per ISA wave32 VGPR layout ----
    V16 af[4], bfr[4];
    #pragma unroll
    for (int i = 0; i < 4; ++i) {
      int m = wm * 64 + i * 16 + row16;
      const bf16* p = lA + m * 40 + khalf * 8;   // K = khalf*8+e / 16+khalf*8+e
      af[i].h[0] = *(const v8bf*)p;
      af[i].h[1] = *(const v8bf*)(p + 16);
    }
    #pragma unroll
    for (int j = 0; j < 4; ++j) {
      int n = wn * 64 + j * 16 + row16;
      const bf16* p = lB + n * 40 + khalf * 8;   // K = khalf*8+e / 16+khalf*8+e
      bfr[j].h[0] = *(const v8bf*)p;
      bfr[j].h[1] = *(const v8bf*)(p + 16);
    }
    #pragma unroll
    for (int i = 0; i < 4; ++i)
      #pragma unroll
      for (int j = 0; j < 4; ++j)
        acc[i][j] = __builtin_amdgcn_wmma_f32_16x16x32_bf16(
            false, af[i].v, false, bfr[j].v, (short)0, acc[i][j], false, false);
  }
  __syncthreads();

  // epilogue: lane l -> N = nbase + l%16, M = mbase + r + (l/16)*8
  #pragma unroll
  for (int i = 0; i < 4; ++i) {
    #pragma unroll
    for (int j = 0; j < 4; ++j) {
      int col = nTile + wn * 64 + j * 16 + row16;
      float bv = bias[col];
      #pragma unroll
      for (int r = 0; r < 8; ++r) {
        int row = mTile + wm * 64 + i * 16 + r + khalf * 8;
        float v = acc[i][j][r] + bv;
        if (relu) v = fmaxf(v, 0.f);
        C[(size_t)row * N + col] = (bf16)v;
      }
    }
  }
}

// ---------------------------------------------------------------------------
// 4) classifier / regressor heads
// ---------------------------------------------------------------------------
__global__ void heads_kernel(const bf16* __restrict__ h2,
    const float* __restrict__ Wc, const float* __restrict__ bc,
    const float* __restrict__ Wr, const float* __restrict__ br,
    float* __restrict__ logits)
{
  int idx = blockIdx.x * blockDim.x + threadIdx.x;
  if (idx >= BATCH * RKEEP * 105) return;
  int j = idx % 105, m = idx / 105;
  const bf16* hr = h2 + (size_t)m * HID;
  float sum;
  if (j < 21) {
    sum = bc[j];
    for (int k = 0; k < HID; ++k) sum += (float)hr[k] * Wc[(size_t)k * 21 + j];
  } else {
    int q = j - 21;
    sum = br[q];
    for (int k = 0; k < HID; ++k) sum += (float)hr[k] * Wr[(size_t)k * 84 + q];
  }
  logits[(size_t)m * 105 + j] = sum;
}

// ---------------------------------------------------------------------------
// 5) softmax + bbox decode
// ---------------------------------------------------------------------------
__global__ void softmax_decode_kernel(const float* __restrict__ logits,
    const float* __restrict__ rois, float* __restrict__ prob,
    float* __restrict__ dbox)
{
  int m = blockIdx.x * blockDim.x + threadIdx.x;
  if (m >= BATCH * RKEEP) return;
  const float* L = logits + (size_t)m * 105;
  float mx = L[0];
  for (int c = 1; c < NCLS; ++c) mx = fmaxf(mx, L[c]);
  float e[NCLS], sum = 0.f;
  for (int c = 0; c < NCLS; ++c) { e[c] = expf(L[c] - mx); sum += e[c]; }
  float inv = 1.f / sum;
  for (int c = 0; c < NCLS; ++c) prob[(size_t)m * NCLS + c] = e[c] * inv;

  const float* R = rois + (size_t)m * 4;
  float px = R[0], py = R[1], pw = R[2], ph = R[3];
  for (int c = 0; c < NCLS; ++c) {
    float dx = L[21 + c*4 + 0], dy = L[21 + c*4 + 1];
    float dw = L[21 + c*4 + 2], dh = L[21 + c*4 + 3];
    float nx = px + dx * pw, ny = py + dy * ph;
    float nw = pw * expf(fminf(fmaxf(dw, -10.f), 10.f));
    float nh = ph * expf(fminf(fmaxf(dh, -10.f), 10.f));
    float* o = dbox + ((size_t)m * NCLS + c) * 4;
    o[0] = nx; o[1] = ny; o[2] = nx + nw; o[3] = ny + nh;
  }
}

// ---------------------------------------------------------------------------
// 6) per-(image,class) NMS
// ---------------------------------------------------------------------------
__global__ __launch_bounds__(512) void class_nms_kernel(
    const float* __restrict__ prob, const int* __restrict__ roi_valid,
    const float* __restrict__ dbox,
    float* __restrict__ fb, float* __restrict__ fs)
{
  __shared__ float ss[RKEEP];
  __shared__ int   si[RKEEP];
  __shared__ float bx[RKEEP][4];
  __shared__ unsigned char keep[RKEEP];

  const int img = blockIdx.x / 20;
  const int cls = blockIdx.x % 20 + 1;
  const int tid = threadIdx.x;
  const int m   = img * RKEEP + tid;

  float sc = prob[(size_t)m * NCLS + cls];
  bool v = (sc > 0.05f) && (roi_valid[m] != 0);
  ss[tid] = v ? sc : NEGV;
  si[tid] = tid;
  __syncthreads();

  for (int k = 2; k <= RKEEP; k <<= 1)
    for (int j = k >> 1; j > 0; j >>= 1) {
      int p = tid ^ j;
      if (p > tid) {
        bool dir = ((tid & k) == 0);
        float a = ss[tid], b = ss[p];
        bool sw = dir ? (a < b) : (a > b);
        if (sw) { ss[tid] = b; ss[p] = a; int t = si[tid]; si[tid] = si[p]; si[p] = t; }
      }
      __syncthreads();
    }

  int o = si[tid];
  const float* db = dbox + (((size_t)(img * RKEEP + o)) * NCLS + cls) * 4;
  bx[tid][0] = db[0]; bx[tid][1] = db[1]; bx[tid][2] = db[2]; bx[tid][3] = db[3];
  keep[tid] = (ss[tid] > NEGV * 0.5f) ? 1 : 0;
  __syncthreads();

  for (int i = 0; i < RKEEP; ++i) {
    if (i < tid && keep[i]) {
      float u1 = fmaxf(bx[i][0], bx[tid][0]), v1 = fmaxf(bx[i][1], bx[tid][1]);
      float u2 = fminf(bx[i][2], bx[tid][2]), v2 = fminf(bx[i][3], bx[tid][3]);
      float iw = fmaxf(u2 - u1, 0.f), ih = fmaxf(v2 - v1, 0.f);
      float inter = iw * ih;
      float ai = (bx[i][2] - bx[i][0]) * (bx[i][3] - bx[i][1]);
      float aj = (bx[tid][2] - bx[tid][0]) * (bx[tid][3] - bx[tid][1]);
      float iou = inter / fmaxf(ai + aj - inter, 1e-6f);
      if (iou > 0.5f) keep[tid] = 0;
    }
    __syncthreads();
  }

  size_t base = (((size_t)img * 20 + (cls - 1)) * RKEEP + tid);
  float* fbp = fb + base * 4;
  fbp[0] = bx[tid][0]; fbp[1] = bx[tid][1]; fbp[2] = bx[tid][2]; fbp[3] = bx[tid][3];
  fs[base] = keep[tid] ? ss[tid] : NEGV;
}

// ---------------------------------------------------------------------------
// 7) top-300 per image
// ---------------------------------------------------------------------------
__global__ __launch_bounds__(1024) void topk_kernel(
    const float* __restrict__ fb, const float* __restrict__ fs,
    const float* __restrict__ scale_ratio, float* __restrict__ out)
{
  const int NTOT = 20 * RKEEP;  // 10240
  __shared__ float cs[20 * RKEEP];
  __shared__ float rs[1024];
  __shared__ int   ri[1024];

  const int img = blockIdx.x, tid = threadIdx.x;
  for (int i = tid; i < NTOT; i += 1024) cs[i] = fs[(size_t)img * NTOT + i];
  __syncthreads();

  const float inv = 1.f / scale_ratio[img];

  for (int it = 0; it < MAXDET; ++it) {
    float best = -3.0e9f; int bi = NTOT;
    for (int i = tid; i < NTOT; i += 1024) {
      float vv = cs[i];
      if (vv > best) { best = vv; bi = i; }
    }
    rs[tid] = best; ri[tid] = bi;
    __syncthreads();
    for (int s = 512; s > 0; s >>= 1) {
      if (tid < s) {
        float b = rs[tid + s];
        if (b > rs[tid] || (b == rs[tid] && ri[tid + s] < ri[tid])) {
          rs[tid] = b; ri[tid] = ri[tid + s];
        }
      }
      __syncthreads();
    }
    if (tid == 0) {
      int idx = ri[0]; float sc = rs[0];
      float* o = out + ((size_t)img * MAXDET + it) * 6;
      bool valid = (sc > NEGV * 0.5f) && (idx < NTOT);
      if (valid) {
        const float* b = fb + ((size_t)img * NTOT + idx) * 4;
        o[0] = b[0] * inv; o[1] = b[1] * inv; o[2] = b[2] * inv; o[3] = b[3] * inv;
        o[4] = sc * inv;
        o[5] = (float)(idx / RKEEP + 1);
      } else {
        o[0] = o[1] = o[2] = o[3] = o[4] = 0.f; o[5] = -1.f;
      }
      if (idx < NTOT) cs[idx] = -3.0e9f;
    }
    __syncthreads();
  }
}

// ---------------------------------------------------------------------------
extern "C" void kernel_launch(void* const* d_in, const int* in_sizes, int n_in,
                              void* d_out, int out_size, void* d_ws, size_t ws_size,
                              hipStream_t stream)
{
  (void)in_sizes; (void)n_in; (void)out_size; (void)ws_size;

  const float* feat      = (const float*)d_in[0];
  const float* proposals = (const float*)d_in[1];
  const float* obj       = (const float*)d_in[2];
  const float* scale     = (const float*)d_in[3];
  const float* W1        = (const float*)d_in[4];
  const float* b1        = (const float*)d_in[5];
  const float* W2        = (const float*)d_in[6];
  const float* b2        = (const float*)d_in[7];
  const float* Wc        = (const float*)d_in[8];
  const float* bc        = (const float*)d_in[9];
  const float* Wr        = (const float*)d_in[10];
  const float* br        = (const float*)d_in[11];

  char* ws = (char*)d_ws;
  size_t off = 0;
  auto alloc = [&](size_t bytes) -> void* {
    off = (off + 255) & ~(size_t)255;
    void* p = ws + off;
    off += bytes;
    return p;
  };

  float* rois      = (float*)alloc((size_t)BATCH * RKEEP * 4 * sizeof(float));
  int*   roi_valid = (int*)  alloc((size_t)BATCH * RKEEP * sizeof(int));
  bf16*  pooled    = (bf16*) alloc((size_t)BATCH * RKEEP * DPOOL * sizeof(bf16));
  bf16*  h1        = (bf16*) alloc((size_t)BATCH * RKEEP * HID * sizeof(bf16));
  bf16*  h2        = (bf16*) alloc((size_t)BATCH * RKEEP * HID * sizeof(bf16));
  float* logits    = (float*)alloc((size_t)BATCH * RKEEP * 105 * sizeof(float));
  float* prob      = (float*)alloc((size_t)BATCH * RKEEP * NCLS * sizeof(float));
  float* dbox      = (float*)alloc((size_t)BATCH * RKEEP * NCLS * 4 * sizeof(float));
  float* fbv       = (float*)alloc((size_t)BATCH * 20 * RKEEP * 4 * sizeof(float));
  float* fsv       = (float*)alloc((size_t)BATCH * 20 * RKEEP * sizeof(float));

  rpn_nms_kernel<<<BATCH, 1024, 0, stream>>>(proposals, obj, rois, roi_valid);

  roi_pool_kernel<<<(BATCH * RKEEP * CH + 255) / 256, 256, 0, stream>>>(
      feat, rois, pooled);

  // FC1: [1024 x 25088] x [25088 x 4096]
  gemm_bf16_kernel<<<dim3(HID / 256, MROWS / 128), 256, 0, stream>>>(
      pooled, W1, b1, h1, DPOOL, HID, 1);

  // FC2: [1024 x 4096] x [4096 x 4096]
  gemm_bf16_kernel<<<dim3(HID / 256, MROWS / 128), 256, 0, stream>>>(
      h1, W2, b2, h2, HID, HID, 1);

  heads_kernel<<<(BATCH * RKEEP * 105 + 255) / 256, 256, 0, stream>>>(
      h2, Wc, bc, Wr, br, logits);

  softmax_decode_kernel<<<(BATCH * RKEEP + 255) / 256, 256, 0, stream>>>(
      logits, rois, prob, dbox);

  class_nms_kernel<<<BATCH * 20, RKEEP, 0, stream>>>(
      prob, roi_valid, dbox, fbv, fsv);

  topk_kernel<<<BATCH, 1024, 0, stream>>>(fbv, fsv, scale, (float*)d_out);
}
